// GlobalOnlyMultiHeadAttention_62612033241273
// MI455X (gfx1250) — compile-verified, matrix operand order checked by reading, not run on hardware
//
#include <hip/hip_runtime.h>
#include <hip/hip_bf16.h>

typedef __attribute__((ext_vector_type(16))) _Float16 v16h;
typedef __attribute__((ext_vector_type(8)))  float    v8f;

#define DEV __device__ __forceinline__

// ---------------- problem constants ----------------
#define Bb 4
#define Nn 384
#define Ll 64
#define Dd 256
#define Hh 8
#define Hd 32
#define Pp 4

// ---------------- workspace layout (bytes) ----------------
static constexpr size_t OFF_SUMF = 0;                          // f32 [1536][256]
static constexpr size_t OFF_SUMH = OFF_SUMF + 1536ull*256*4;   // f16 [1536][256]
static constexpr size_t OFF_PS   = OFF_SUMH + 1536ull*256*2;   // f32 [1536][4]
static constexpr size_t OFF_WQH  = OFF_PS   + 1536ull*4*4;     // f16 [256][256]
static constexpr size_t OFF_WKH  = OFF_WQH  + 65536ull*2;
static constexpr size_t OFF_WVH  = OFF_WKH  + 65536ull*2;
static constexpr size_t OFF_WGH  = OFF_WVH  + 65536ull*2;      // f16 [256][512]
static constexpr size_t OFF_QH   = OFF_WGH  + 131072ull*2;     // f16 [b][h][n][d]
static constexpr size_t OFF_KH   = OFF_QH   + 393216ull*2;
static constexpr size_t OFF_VTH  = OFF_KH   + 393216ull*2;     // f16 [b][h][d][n]
static constexpr size_t OFF_USF  = OFF_VTH  + 393216ull*2;     // f32 [1536][256]

static constexpr size_t OUT_US   = (size_t)Bb*Nn*Ll*Dd;        // d_out offset of updated_summary

// ---------------- helpers ----------------
DEV float wave_sum(float v) {
#pragma unroll
  for (int m = 16; m >= 1; m >>= 1) v += __shfl_xor(v, m, 32);
  return v;
}
DEV float wave_max(float v) {
#pragma unroll
  for (int m = 16; m >= 1; m >>= 1) v = fmaxf(v, __shfl_xor(v, m, 32));
  return v;
}
DEV float gelu_exact(float x) { return 0.5f * x * (1.0f + erff(x * 0.70710678118654752f)); }

// Row-major 16x32 f16 fragment (A, or B as row-major B^T), CDNA5 wave32 layout:
// lanes 0-15: rows 0-15, halves[0..7]=K 0..7, halves[8..15]=K 16..23
// lanes 16-31: rows 0-15, halves[0..7]=K 8..15, halves[8..15]=K 24..31
DEV v16h frag_load_rm(const _Float16* base, int stride) {
  int lane = threadIdx.x & 31;
  const _Float16* p = base + (lane & 15) * stride + ((lane >> 4) << 3);
  v16h f;
#pragma unroll
  for (int i = 0; i < 8; ++i) { f[i] = p[i]; f[i + 8] = p[i + 16]; }
  return f;
}
// Same, but clamp row index to [0,7] (for M=8 padded to 16 tiles: rows 8..15 duplicate row 7, results ignored)
DEV v16h frag_load_rm_clamp8(const _Float16* base, int stride) {
  int lane = threadIdx.x & 31;
  int r = lane & 15; if (r > 7) r = 7;
  const _Float16* p = base + r * stride + ((lane >> 4) << 3);
  v16h f;
#pragma unroll
  for (int i = 0; i < 8; ++i) { f[i] = p[i]; f[i + 8] = p[i + 16]; }
  return f;
}
DEV v8f wmma_f16(v16h a, v16h b, v8f c) {
  return __builtin_amdgcn_wmma_f32_16x16x32_f16(false, a, false, b, (short)0, c, false, false);
}

// ================= Kernel A: masked mean summary + positions mean =================
__global__ void k_summarize(const float* __restrict__ x, const float* __restrict__ mask,
                            const float* __restrict__ pos,
                            float* __restrict__ sumf, _Float16* __restrict__ sumh,
                            float* __restrict__ psout) {
  int t = blockIdx.x;                // b*N + n
  int c = threadIdx.x;               // 0..255 channel
  const float* mrow = mask + (size_t)t * Ll;
  float wsum = 0.f;
#pragma unroll 8
  for (int l = 0; l < Ll; ++l) wsum += mrow[l];
  float denom = fmaxf(wsum, 1e-6f);
  const float* xrow = x + (size_t)t * Ll * Dd;
  float s = 0.f;
#pragma unroll 4
  for (int l = 0; l < Ll; ++l) s += xrow[(size_t)l * Dd + c] * mrow[l];
  float sm = s / denom;
  sumf[(size_t)t * Dd + c] = sm;
  sumh[(size_t)t * Dd + c] = (_Float16)sm;
  if (c < Pp) {
    const float* prow = pos + (size_t)t * Ll * Pp;
    float p = 0.f;
    for (int l = 0; l < Ll; ++l) p += prow[l * Pp + c] * mrow[l];
    psout[(size_t)t * Pp + c] = p / denom;
  }
}

// ================= Kernel W: fp32 -> f16 convert =================
__global__ void k_cvt(const float* __restrict__ s, _Float16* __restrict__ d, int n) {
  int i = blockIdx.x * 256 + threadIdx.x;
  if (i < n) d[i] = (_Float16)s[i];
}

// ================= Kernel B: QKV projections (WMMA) =================
__global__ void k_qkv(const _Float16* __restrict__ sumh,
                      const _Float16* __restrict__ Wqh, const _Float16* __restrict__ Wkh,
                      const _Float16* __restrict__ Wvh,
                      const float* __restrict__ bq, const float* __restrict__ bk,
                      const float* __restrict__ bv,
                      _Float16* __restrict__ qh, _Float16* __restrict__ kh,
                      _Float16* __restrict__ vTh) {
  int mt   = blockIdx.x;             // 0..95 row-tile (16 rows of 1536)
  int widx = blockIdx.y;             // 0=q 1=k 2=v
  int wave = threadIdx.x >> 5, lane = threadIdx.x & 31;
  const _Float16* W = (widx == 0) ? Wqh : (widx == 1) ? Wkh : Wvh;
  const float* bias = (widx == 0) ? bq : (widx == 1) ? bk : bv;
  int row0 = mt * 16;
  for (int s = 0; s < 2; ++s) {
    int nt = wave * 2 + s;           // 0..15 col-tile
    v8f acc = {};
#pragma unroll
    for (int kc = 0; kc < 8; ++kc) {
      v16h a  = frag_load_rm(sumh + (size_t)row0 * Dd + kc * 32, Dd);
      v16h bf = frag_load_rm(W    + (size_t)nt * 16 * Dd + kc * 32, Dd);
      acc = wmma_f16(a, bf, acc);
    }
    int n = lane & 15, rbase = (lane >> 4) * 8;
    int cc = nt * 16 + n;
    float bsv = bias[cc];
    int h = cc >> 5, d = cc & 31;
#pragma unroll
    for (int v = 0; v < 8; ++v) {
      int row = row0 + rbase + v;
      int b_ = row / Nn, ntok = row % Nn;
      float val = acc[v] + bsv;
      if (widx == 0)      qh [((size_t)(b_ * Hh + h) * Nn + ntok) * Hd + d] = (_Float16)val;
      else if (widx == 1) kh [((size_t)(b_ * Hh + h) * Nn + ntok) * Hd + d] = (_Float16)val;
      else                vTh[(((size_t)b_ * Hh + h) * Hd + d) * Nn + ntok] = (_Float16)val;
    }
  }
}

// ================= Kernel C: attention core =================
// One workgroup per (b, i-tile of 16). 256 threads = 8 waves.
// LDS layout (dynamic):
static constexpr int AW_OFF  = 0;                    // f16 [8][16][384] = 98304 B (rb bias, then aw)
static constexpr int LG_OFF  = AW_OFF  + 98304;      // f32 [16][384]   = 24576 B (logit/exp scratch)
static constexpr int GT_OFF  = LG_OFF  + 24576;      // per-wave 8192 B: f16 [128][32] g-tile / f32 [16][128] Gh dump
static constexpr int A1_OFF  = GT_OFF  + 65536;      // f32 [16][256]   = 16384 B (attn1)
static constexpr int PSJ_OFF = A1_OFF  + 16384;      // f32 [384][4]    = 6144 B
static constexpr int PSI_OFF = PSJ_OFF + 6144;       // f32 [16][4]     = 256 B
static constexpr int WB_OFF  = PSI_OFF + 256;        // f32 weights     = 7456 B
static constexpr int SMEM_C  = WB_OFF  + 7456;       // 218656 B total

__global__ void k_attn(const _Float16* __restrict__ qh, const _Float16* __restrict__ kh,
                       const _Float16* __restrict__ vTh,
                       const float* __restrict__ ps, const float* __restrict__ sumf,
                       const float* __restrict__ rb_w1, const float* __restrict__ rb_b1,
                       const float* __restrict__ rb_w2, const float* __restrict__ rb_b2,
                       const float* __restrict__ rv_w1, const float* __restrict__ rv_b1,
                       const float* __restrict__ rv_w2, const float* __restrict__ rv_b2,
                       float* __restrict__ usf, float* __restrict__ out_us) {
  extern __shared__ char smem[];
  _Float16* aw    = (_Float16*)(smem + AW_OFF);
  float*    lgt   = (float*)(smem + LG_OFF);
  float*    attn1 = (float*)(smem + A1_OFF);
  float*    psj   = (float*)(smem + PSJ_OFF);
  float*    psi   = (float*)(smem + PSI_OFF);
  float*    wb    = (float*)(smem + WB_OFF);
  float* rbw1s = wb;           float* rbb1s = wb + 384;  float* rbw2s = wb + 448;
  float* rbb2s = wb + 960;     float* rvw1s = wb + 968;  float* rvb1s = wb + 1736;

  int tid = threadIdx.x, wave = tid >> 5, lane = tid & 31;
  int b = blockIdx.x / (Nn / 16);
  int ibase = (blockIdx.x % (Nn / 16)) * 16;

  // ---- phase 0: stage ps and small weights ----
  for (int i = tid; i < Nn * Pp; i += 256) psj[i] = ps[(size_t)b * Nn * Pp + i];
  for (int i = tid; i < 16 * Pp; i += 256) psi[i] = ps[((size_t)b * Nn + ibase) * Pp + i];
  for (int i = tid; i < 384; i += 256) rbw1s[i] = rb_w1[i];
  for (int i = tid; i < 64;  i += 256) rbb1s[i] = rb_b1[i];
  for (int i = tid; i < 512; i += 256) rbw2s[i] = rb_w2[i];
  for (int i = tid; i < 8;   i += 256) rbb2s[i] = rb_b2[i];
  for (int i = tid; i < 768; i += 256) rvw1s[i] = rv_w1[i];
  for (int i = tid; i < 128; i += 256) rvb1s[i] = rv_b1[i];
  __syncthreads();

  // ---- phase 1: relative attention bias (6->64 gelu ->8) stored into aw as f16 ----
  for (int p = tid; p < 16 * Nn; p += 256) {
    int i = p / Nn, j = p % Nn;
    float f6[6];
    float d2 = 0.f;
#pragma unroll
    for (int k = 0; k < 4; ++k) { float d = psi[i * 4 + k] - psj[j * 4 + k]; f6[k] = d; d2 += d * d; }
    f6[4] = sqrtf(d2); f6[5] = d2;
    float o[8];
#pragma unroll
    for (int h = 0; h < 8; ++h) o[h] = rbb2s[h];
    for (int m = 0; m < 64; ++m) {
      float hid = rbb1s[m];
#pragma unroll
      for (int k = 0; k < 6; ++k) hid += f6[k] * rbw1s[m * 6 + k];
      float g = gelu_exact(hid);
#pragma unroll
      for (int h = 0; h < 8; ++h) o[h] += g * rbw2s[h * 64 + m];
    }
#pragma unroll
    for (int h = 0; h < 8; ++h) aw[((size_t)h * 16 + i) * Nn + j] = (_Float16)o[h];
  }
  __syncthreads();

  // ---- phase 2: logits (WMMA, K=Hd=32) + softmax, per head ----
  const float scale = 0.17677669529663689f;   // 1/sqrt(32)
  for (int h = 0; h < Hh; ++h) {
    __syncthreads();   // logits scratch reuse across heads
    v16h afr = frag_load_rm(qh + ((size_t)(b * Hh + h) * Nn + ibase) * Hd, Hd);
#pragma unroll
    for (int s = 0; s < 3; ++s) {
      int jt = wave + s * 8;                  // 24 j-tiles / 8 waves
      v16h bfr = frag_load_rm(kh + ((size_t)(b * Hh + h) * Nn + jt * 16) * Hd, Hd);
      v8f c = {};
      c = wmma_f16(afr, bfr, c);
      int n = lane & 15, rbse = (lane >> 4) * 8;
#pragma unroll
      for (int v = 0; v < 8; ++v) lgt[(rbse + v) * Nn + jt * 16 + n] = c[v] * scale;
    }
    __syncthreads();
    // softmax: 2 rows per wave
#pragma unroll
    for (int s = 0; s < 2; ++s) {
      int r = wave * 2 + s;
      float* lrow = lgt + (size_t)r * Nn;
      _Float16* arow = aw + ((size_t)h * 16 + r) * Nn;
      float mx = -1e30f;
      for (int j = lane; j < Nn; j += 32) mx = fmaxf(mx, lrow[j] + (float)arow[j]);
      mx = wave_max(mx);
      float sum = 0.f;
      for (int j = lane; j < Nn; j += 32) {
        float e = __expf(lrow[j] + (float)arow[j] - mx);
        lrow[j] = e; sum += e;
      }
      sum = wave_sum(sum);
      float inv = 1.f / sum;
      for (int j = lane; j < Nn; j += 32) arow[j] = (_Float16)(lrow[j] * inv);
    }
  }
  __syncthreads();

  // ---- phase 3: attn1 = aw @ v  (WMMA over j, vT layout gives contiguous B frags) ----
#pragma unroll
  for (int s = 0; s < 2; ++s) {
    int task = wave * 2 + s;                  // 16 tasks: (h, d-half)
    int h = task >> 1, dp = task & 1;
    v8f acc = {};
    for (int jc = 0; jc < Nn / 32; ++jc) {
      v16h afr = frag_load_rm(aw + (size_t)(h * 16) * Nn + jc * 32, Nn);
      v16h bfr = frag_load_rm(vTh + (((size_t)(b * Hh + h)) * Hd + dp * 16) * Nn + jc * 32, Nn);
      acc = wmma_f16(afr, bfr, acc);
    }
    int n = lane & 15, rbse = (lane >> 4) * 8;
#pragma unroll
    for (int v = 0; v < 8; ++v) attn1[(rbse + v) * Dd + h * Hd + dp * 16 + n] = acc[v];
  }
  __syncthreads();

  // ---- phase 4: Gh[h,m] = sum_j aw[h,i,j]*gelu(rv1(feat))[j,m]  then  attn2 = Gh . W2 ----
  _Float16* gt  = (_Float16*)(smem + GT_OFF + wave * 8192);   // f16 [128][32]  (this wave)
  float*    ghb = (float*)   (smem + GT_OFF + wave * 8192);   // f32 [16][128]  (reused)
  for (int ii = 0; ii < 2; ++ii) {
    int i = wave * 2 + ii;                    // local row within the 16-row tile
    float fx[4];
#pragma unroll
    for (int k = 0; k < 4; ++k) fx[k] = psi[i * 4 + k];
    v8f gh[8] = {};
    for (int jc = 0; jc < Nn / 32; ++jc) {
      int jg = jc * 32 + lane;
      float f6[6]; float d2 = 0.f;
#pragma unroll
      for (int k = 0; k < 4; ++k) { float d = fx[k] - psj[jg * 4 + k]; f6[k] = d; d2 += d * d; }
      f6[4] = sqrtf(d2); f6[5] = d2;
      for (int m = 0; m < 128; ++m) {
        float hid = rvb1s[m];
#pragma unroll
        for (int k = 0; k < 6; ++k) hid += f6[k] * rvw1s[m * 6 + k];
        gt[m * 32 + lane] = (_Float16)gelu_exact(hid);
      }
      __syncthreads();   // uniform across all waves
      v16h afr = frag_load_rm_clamp8(aw + (size_t)i * Nn + jc * 32, 16 * Nn);  // rows = h
#pragma unroll
      for (int nt = 0; nt < 8; ++nt) {
        v16h bfr = frag_load_rm(gt + nt * 16 * 32, 32);
        gh[nt] = wmma_f16(afr, bfr, gh[nt]);
      }
      __syncthreads();   // before g-tile overwrite
    }
    // dump Gh rows 0..7 (valid heads) to LDS
    {
      int n = lane & 15;
      if (lane < 16) {
#pragma unroll
        for (int nt = 0; nt < 8; ++nt)
#pragma unroll
          for (int v = 0; v < 8; ++v) ghb[v * 128 + nt * 16 + n] = gh[nt][v];
      }
    }
    __syncthreads();
    // attn2[c] = rv_b2[c] (softmax sums to 1) + Gh[h(c),:] . W2[c,:]; combine and emit updated_summary
    size_t grow = ((size_t)b * Nn + ibase + i) * Dd;
    for (int c = lane; c < Dd; c += 32) {
      int h = c >> 5;
      float dot = rv_b2[c];
      const float* w2r = rv_w2 + (size_t)c * 128;
      const float* gr  = ghb + h * 128;
#pragma unroll 8
      for (int m = 0; m < 128; ++m) dot += gr[m] * w2r[m];
      float us = dot + attn1[i * Dd + c] + sumf[grow + c];
      usf[grow + c] = us;
      out_us[grow + c] = us;
    }
    __syncthreads();
  }
}

// ================= Kernel D: normalization + gate GEMM (WMMA) + output =================
static constexpr int SMEM_D = 64 * 512 * 2 /*comb f16*/ + 64 * 256 * 4 /*pregate f32*/ + 256 * 4 /*us row*/;

__global__ void k_gate(const float* __restrict__ x, const float* __restrict__ usfin,
                       const _Float16* __restrict__ Wgh, const float* __restrict__ bg,
                       const float* __restrict__ gng, const float* __restrict__ gnb,
                       float* __restrict__ out) {
  extern __shared__ char smem[];
  _Float16* comb    = (_Float16*)smem;                       // [64][512]
  float*    pregate = (float*)(smem + 65536);                // [64][256]
  float*    usl     = (float*)(smem + 65536 + 65536);        // [256]
  int t = blockIdx.x;                // b*N + n
  int tid = threadIdx.x, wave = tid >> 5, lane = tid & 31;

  usl[tid] = usfin[(size_t)t * Dd + tid];
  __syncthreads();

  // global-norm of updated summary (redundant per wave, no extra sync)
  float gsq = 0.f;
#pragma unroll
  for (int k = 0; k < 8; ++k) { float u = usl[lane + 32 * k]; gsq += u * u; }
  gsq = wave_sum(gsq);
  float ginv = 1.f / fmaxf(sqrtf(gsq), 1e-6f);

  const float* xblk = x + (size_t)t * Ll * Dd;
  // per-row local norm + build combined [local_n | global_n] in f16
  for (int rr = 0; rr < 8; ++rr) {
    int l = wave * 8 + rr;
    const float* xr = xblk + (size_t)l * Dd;
    float ls = 0.f;
#pragma unroll
    for (int k = 0; k < 8; ++k) { float xv = xr[lane + 32 * k]; ls += xv * xv; }
    ls = wave_sum(ls);
    float linv = 1.f / fmaxf(sqrtf(ls), 1e-6f);
#pragma unroll
    for (int k = 0; k < 8; ++k) {
      int c = lane + 32 * k;
      comb[(size_t)l * 512 + c]       = (_Float16)(xr[c] * linv);
      comb[(size_t)l * 512 + 256 + c] = (_Float16)(usl[c] * ginv);
    }
  }
  __syncthreads();

  // gate GEMM: (64 x 512) @ (512 x 256), 64 tiles over 8 waves, K in 16 chunks
#pragma unroll
  for (int s = 0; s < 8; ++s) {
    int task = wave * 8 + s;
    int mt = task >> 4, nt = task & 15;
    v8f acc = {};
    for (int kc = 0; kc < 16; ++kc) {
      v16h afr = frag_load_rm(comb + (size_t)mt * 16 * 512 + kc * 32, 512);
      v16h bfr = frag_load_rm(Wgh + (size_t)nt * 16 * 512 + kc * 32, 512);
      acc = wmma_f16(afr, bfr, acc);
    }
    int n = lane & 15, rbse = (lane >> 4) * 8;
    int c = nt * 16 + n;
    float bgv = bg[c];
#pragma unroll
    for (int v = 0; v < 8; ++v) {
      float pv = acc[v] + bgv;
      pregate[(size_t)(mt * 16 + rbse + v) * Dd + c] = 1.f / (1.f + __expf(-pv));
    }
  }
  __syncthreads();

  // per-row LayerNorm of the sigmoid gate, then blend
  for (int rr = 0; rr < 8; ++rr) {
    int l = wave * 8 + rr;
    float* pr = pregate + (size_t)l * Dd;
    float s = 0.f;
#pragma unroll
    for (int k = 0; k < 8; ++k) s += pr[lane + 32 * k];
    s = wave_sum(s);
    float mean = s * (1.f / 256.f);
    float vs = 0.f;
#pragma unroll
    for (int k = 0; k < 8; ++k) { float d = pr[lane + 32 * k] - mean; vs += d * d; }
    vs = wave_sum(vs);
    float rstd = rsqrtf(vs * (1.f / 256.f) + 1e-5f);
    const float* xr = xblk + (size_t)l * Dd;
    float* orow = out + ((size_t)t * Ll + l) * Dd;
#pragma unroll
    for (int k = 0; k < 8; ++k) {
      int c = lane + 32 * k;
      float gate = (pr[c] - mean) * rstd * gng[c] + gnb[c];
      orow[c] = (1.f - gate) * xr[c] + gate * usl[c];
    }
  }
}

// ================= host launch =================
extern "C" void kernel_launch(void* const* d_in, const int* in_sizes, int n_in,
                              void* d_out, int out_size, void* d_ws, size_t ws_size,
                              hipStream_t stream) {
  const float* x     = (const float*)d_in[0];
  // d_in[1] sum_token unused (zeros)
  const float* masks = (const float*)d_in[2];
  const float* pos   = (const float*)d_in[3];
  const float* Wq = (const float*)d_in[4];   const float* bq = (const float*)d_in[5];
  const float* Wk = (const float*)d_in[6];   const float* bk = (const float*)d_in[7];
  const float* Wv = (const float*)d_in[8];   const float* bv = (const float*)d_in[9];
  const float* Wg = (const float*)d_in[10];  const float* bg = (const float*)d_in[11];
  const float* gng = (const float*)d_in[12]; const float* gnb = (const float*)d_in[13];
  const float* rb_w1 = (const float*)d_in[14]; const float* rb_b1 = (const float*)d_in[15];
  const float* rb_w2 = (const float*)d_in[16]; const float* rb_b2 = (const float*)d_in[17];
  const float* rv_w1 = (const float*)d_in[18]; const float* rv_b1 = (const float*)d_in[19];
  const float* rv_w2 = (const float*)d_in[20]; const float* rv_b2 = (const float*)d_in[21];

  char* ws = (char*)d_ws;
  float*    sumf = (float*)(ws + OFF_SUMF);
  _Float16* sumh = (_Float16*)(ws + OFF_SUMH);
  float*    psb  = (float*)(ws + OFF_PS);
  _Float16* Wqh  = (_Float16*)(ws + OFF_WQH);
  _Float16* Wkh  = (_Float16*)(ws + OFF_WKH);
  _Float16* Wvh  = (_Float16*)(ws + OFF_WVH);
  _Float16* Wgh  = (_Float16*)(ws + OFF_WGH);
  _Float16* qh   = (_Float16*)(ws + OFF_QH);
  _Float16* kh   = (_Float16*)(ws + OFF_KH);
  _Float16* vTh  = (_Float16*)(ws + OFF_VTH);
  float*    usf  = (float*)(ws + OFF_USF);

  float* out    = (float*)d_out;
  float* out_us = out + OUT_US;

  (void)hipFuncSetAttribute((const void*)k_attn, hipFuncAttributeMaxDynamicSharedMemorySize, SMEM_C);
  (void)hipFuncSetAttribute((const void*)k_gate, hipFuncAttributeMaxDynamicSharedMemorySize, SMEM_D);

  // A: summary + positions
  k_summarize<<<Bb * Nn, 256, 0, stream>>>(x, masks, pos, sumf, sumh, psb);
  // W: weight conversions
  k_cvt<<<(65536 + 255) / 256, 256, 0, stream>>>(Wq, Wqh, 65536);
  k_cvt<<<(65536 + 255) / 256, 256, 0, stream>>>(Wk, Wkh, 65536);
  k_cvt<<<(65536 + 255) / 256, 256, 0, stream>>>(Wv, Wvh, 65536);
  k_cvt<<<(131072 + 255) / 256, 256, 0, stream>>>(Wg, Wgh, 131072);
  // B: QKV projection GEMMs
  dim3 gB(96, 3);
  k_qkv<<<gB, 256, 0, stream>>>(sumh, Wqh, Wkh, Wvh, bq, bk, bv, qh, kh, vTh);
  // C: attention core
  k_attn<<<Bb * (Nn / 16), 256, SMEM_C, stream>>>(qh, kh, vTh, psb, sumf,
                                                  rb_w1, rb_b1, rb_w2, rb_b2,
                                                  rv_w1, rv_b1, rv_w2, rv_b2,
                                                  usf, out_us);
  // D: gating + output
  k_gate<<<Bb * Nn, 256, SMEM_D, stream>>>(x, usf, Wgh, bg, gng, gnb, out);
}
